// MoEGate_50749333570099
// MI455X (gfx1250) — matrix operand order repaired and checked by reading, not compiled, not run on hardware
//
#include <hip/hip_runtime.h>
#include <hip/hip_bf16.h>

typedef __attribute__((ext_vector_type(16))) __bf16        v16bf;
typedef __attribute__((ext_vector_type(8)))  float         v8f;
typedef __attribute__((ext_vector_type(4)))  float         v4f;
typedef __attribute__((ext_vector_type(4)))  unsigned int  v4u;

#define HIDDEN        2048
#define NEXP          64
#define NGROUP        8
#define EPG           8
#define TOPKG         3
#define TOPK          6

#define KCHUNK        128
#define WAVES         8
#define BLOCK         (WAVES * 32)
#define TOK_PER_WAVE  16
#define TOK_PER_BLOCK (WAVES * TOK_PER_WAVE)   // 128 tokens per workgroup

// chunk staged per buffer: [hi|lo][64 experts][128 k] bf16 = 32 KB
#define CHUNK_BYTES   (2 * NEXP * KCHUNK * 2)
#define UNITS16       (CHUNK_BYTES / 16)        // 2048 16-byte units
#define UNITS_PER_TID (UNITS16 / BLOCK)         // 8 async b128 per lane per stage

union BV { v4u u[2]; v16bf v; };

// ---------------------------------------------------------------------------
// One-shot pre-pack: W fp32 -> bf16 hi + bf16 residual in d_ws.
// Layout: ws[2][NEXP][HIDDEN] bf16  (h=0: hi, h=1: lo)
// ---------------------------------------------------------------------------
__global__ __launch_bounds__(256) void pack_weight_kernel(
    const float* __restrict__ w, __bf16* __restrict__ wsHi,
    __bf16* __restrict__ wsLo, int n)
{
    const int i = blockIdx.x * 256 + threadIdx.x;
    if (i < n) {
        float  f = w[i];
        __bf16 h = (__bf16)f;
        wsHi[i] = h;
        wsLo[i] = (__bf16)(f - (float)h);
    }
}

// ---------------------------------------------------------------------------
// Async-DMA one weight chunk (hi+lo) into an LDS buffer.
// unit c: he = c>>4 (h*64+e), k16 = c&15
//   global byte off = he*HIDDEN*2 + kc*2 + k16*16
//   lds    byte off = ldsBase + c*16
// ---------------------------------------------------------------------------
__device__ __forceinline__ void stage_chunk_async(
    const char* __restrict__ wsBase, unsigned ldsBase, int kc, int tid)
{
    #pragma unroll
    for (int i = 0; i < UNITS_PER_TID; ++i) {
        const int c   = tid + i * BLOCK;
        const int he  = c >> 4;
        const int k16 = c & 15;
        const unsigned lds = ldsBase + (unsigned)c * 16u;
        const unsigned long long ga = (unsigned long long)(size_t)
            (wsBase + (size_t)he * (HIDDEN * 2) + (size_t)kc * 2 + (size_t)k16 * 16);
        asm volatile("global_load_async_to_lds_b128 %0, %1, off"
                     :: "v"(lds), "v"(ga) : "memory");
    }
}

__global__ __launch_bounds__(BLOCK) void moe_gate_wmma_kernel(
    const float* __restrict__ x, const char* __restrict__ wsPacked,
    float* __restrict__ out, int tokens)
{
    // double-buffered weight chunk: [buf][hi|lo][expert][k]
    __shared__ __align__(16) __bf16 sW[2][2][NEXP][KCHUNK];        // 64 KB
    // +1 pad -> per-lane epilogue reads are bank-conflict-free
    __shared__ float sLogit[WAVES][TOK_PER_WAVE][NEXP + 1];        // ~33 KB

    const int tid  = threadIdx.x;
    const int wave = tid >> 5;
    const int lane = tid & 31;
    const int half = lane >> 4;       // 0: lanes 0-15, 1: lanes 16-31
    const int l16  = lane & 15;

    const int tokBase = blockIdx.x * TOK_PER_BLOCK + wave * TOK_PER_WAVE;
    int rowTok = tokBase + l16;
    if (rowTok >= tokens) rowTok = tokens - 1;       // clamp (grid is exact)
    const float* __restrict__ xrow = x + (size_t)rowTok * HIDDEN;

    const unsigned ldsBuf0 = (unsigned)(size_t)&sW[0][0][0][0];
    const unsigned ldsBuf1 = (unsigned)(size_t)&sW[1][0][0][0];

    v8f acc[4] = {};   // 4 expert tiles of 16 -> 64 experts

    // prologue: start DMA of chunk 0 into buffer 0
    stage_chunk_async(wsPacked, ldsBuf0, 0, tid);

    int p = 0;
    for (int kc = 0; kc < HIDDEN; kc += KCHUNK) {
        // issue DMA for next chunk into the other buffer, then retire current
        if (kc + KCHUNK < HIDDEN) {
            stage_chunk_async(wsPacked, p ? ldsBuf0 : ldsBuf1, kc + KCHUNK, tid);
            asm volatile("s_wait_asynccnt 8" ::: "memory");  // current buffer landed
        } else {
            asm volatile("s_wait_asynccnt 0" ::: "memory");
        }
        // prefetch next x chunk for this wave's rows
        if (kc + KCHUNK < HIDDEN)
            __builtin_prefetch(xrow + kc + KCHUNK, 0, 1);
        __syncthreads();             // all waves' DMA for current buffer visible

        const __bf16 (* __restrict__ bHiRow)[KCHUNK] = sW[p][0];
        const __bf16 (* __restrict__ bLoRow)[KCHUNK] = sW[p][1];

        // ---- compute: 4 k-steps of 32, 4 expert tiles, 3 WMMAs each --------
        #pragma unroll
        for (int ks = 0; ks < KCHUNK; ks += 32) {
            // A (16x32 bf16): lane<16 holds K 0..7 & 16..23, lane>=16 K 8..15 & 24..31
            const int kb = kc + ks + half * 8;
            v4f f0 = *(const v4f*)(xrow + kb + 0);
            v4f f1 = *(const v4f*)(xrow + kb + 4);
            v4f f2 = *(const v4f*)(xrow + kb + 16);
            v4f f3 = *(const v4f*)(xrow + kb + 20);
            v16bf aHi, aLo;
            #pragma unroll
            for (int j = 0; j < 4; ++j) {
                float  g0 = f0[j], g1 = f1[j], g2 = f2[j], g3 = f3[j];
                __bf16 h0 = (__bf16)g0, h1 = (__bf16)g1;
                __bf16 h2 = (__bf16)g2, h3 = (__bf16)g3;
                aHi[j]      = h0; aHi[4 + j]  = h1;
                aHi[8 + j]  = h2; aHi[12 + j] = h3;
                aLo[j]      = (__bf16)(g0 - (float)h0);
                aLo[4 + j]  = (__bf16)(g1 - (float)h1);
                aLo[8 + j]  = (__bf16)(g2 - (float)h2);
                aLo[12 + j] = (__bf16)(g3 - (float)h3);
            }
            // B (32x16 bf16): lane<16 -> col N=lane, K 0..7 & 16..23; lane>=16 K 8..15 & 24..31
            const int kB = ks + half * 8;
            #pragma unroll
            for (int t = 0; t < 4; ++t) {
                const int e = t * 16 + l16;
                BV bHi, bLo;
                bHi.u[0] = *(const v4u*)&bHiRow[e][kB];
                bHi.u[1] = *(const v4u*)&bHiRow[e][kB + 16];
                bLo.u[0] = *(const v4u*)&bLoRow[e][kB];
                bLo.u[1] = *(const v4u*)&bLoRow[e][kB + 16];
                // logits ~= hi*hi + hi*lo + lo*hi  (~fp32 accuracy)
                acc[t] = __builtin_amdgcn_wmma_f32_16x16x32_bf16(
                    false, aHi, false, bHi.v, (short)0, acc[t], false, false);
                acc[t] = __builtin_amdgcn_wmma_f32_16x16x32_bf16(
                    false, aHi, false, bLo.v, (short)0, acc[t], false, false);
                acc[t] = __builtin_amdgcn_wmma_f32_16x16x32_bf16(
                    false, aLo, false, bHi.v, (short)0, acc[t], false, false);
            }
        }
        __syncthreads();   // all waves done reading buf p before it is re-DMA'd
        p ^= 1;
    }

    // ---- scatter C tiles (16x16 f32 layout) to LDS logits ------------------
    #pragma unroll
    for (int t = 0; t < 4; ++t) {
        #pragma unroll
        for (int r = 0; r < 8; ++r) {
            const int m = half ? (8 + r) : r;
            sLogit[wave][m][t * 16 + l16] = acc[t][r];
        }
    }
    __syncthreads();

    // ---- per-token softmax + group-limited top-k (lanes 0-15, one token each)
    if (lane < 16 && (tokBase + l16) < tokens) {
        float* L = sLogit[wave][l16];

        float mx = -__builtin_inff();
        for (int e = 0; e < NEXP; ++e) mx = fmaxf(mx, L[e]);

        float sum = 0.f;
        float gmax[NGROUP];
        #pragma unroll
        for (int g = 0; g < NGROUP; ++g) {
            float gm = 0.f;
            #pragma unroll
            for (int j = 0; j < EPG; ++j) {
                float v = __expf(L[g * EPG + j] - mx);
                L[g * EPG + j] = v;        // keep unnormalized probs
                sum += v;
                gm = fmaxf(gm, v);          // exp monotonic -> group-max score
            }
            gmax[g] = gm;
        }

        // top-3 groups
        unsigned taken = 0u;
        for (int it = 0; it < TOPKG; ++it) {
            float best = -1.f; int bi = 0;
            #pragma unroll
            for (int g = 0; g < NGROUP; ++g)
                if (!((taken >> g) & 1u) && gmax[g] > best) { best = gmax[g]; bi = g; }
            taken |= 1u << bi;
        }

        // top-6 experts within allowed groups (register insertion sort)
        float tv[TOPK]; int ti[TOPK];
        #pragma unroll
        for (int j = 0; j < TOPK; ++j) { tv[j] = -1.f; ti[j] = 0; }
        for (int e = 0; e < NEXP; ++e) {
            if (!((taken >> (e >> 3)) & 1u)) continue;
            float cv = L[e]; int ci = e;
            #pragma unroll
            for (int j = 0; j < TOPK; ++j) {
                if (cv > tv[j]) {
                    float fv = tv[j]; int fi = ti[j];
                    tv[j] = cv; ti[j] = ci; cv = fv; ci = fi;
                }
            }
        }

        const float inv = 1.f / sum;            // softmax over all 64 experts
        const size_t tok = (size_t)(tokBase + l16);
        float* oIdx = out;
        float* oW   = out + (size_t)tokens * TOPK;
        #pragma unroll
        for (int j = 0; j < TOPK; ++j) {
            oIdx[tok * TOPK + j] = (float)ti[j];
            oW[tok * TOPK + j]   = tv[j] * inv;
        }
    }
}

extern "C" void kernel_launch(void* const* d_in, const int* in_sizes, int n_in,
                              void* d_out, int out_size, void* d_ws, size_t ws_size,
                              hipStream_t stream) {
    const float* x = (const float*)d_in[0];   // [4,4096,2048] fp32
    const float* w = (const float*)d_in[1];   // [64,2048] fp32
    float* out = (float*)d_out;               // [T*6 idx-as-float | T*6 weights]
    const int tokens = in_sizes[0] / HIDDEN;  // 16384

    // d_ws: [2][NEXP][HIDDEN] bf16 = 512 KB (hi plane then lo plane)
    __bf16* wsHi = (__bf16*)d_ws;
    __bf16* wsLo = wsHi + (size_t)NEXP * HIDDEN;
    const int nW = NEXP * HIDDEN;
    pack_weight_kernel<<<(nW + 255) / 256, 256, 0, stream>>>(w, wsHi, wsLo, nW);

    dim3 grid((tokens + TOK_PER_BLOCK - 1) / TOK_PER_BLOCK);
    moe_gate_wmma_kernel<<<grid, BLOCK, 0, stream>>>(
        x, (const char*)d_ws, out, tokens);
}
